// AttentionBlock_17239998726454
// MI455X (gfx1250) — matrix operand order repaired and checked by reading, not compiled, not run on hardware
//
#include <hip/hip_runtime.h>
#include <hip/hip_bf16.h>
#include <math.h>

typedef __attribute__((ext_vector_type(16))) __bf16 v16bf;
typedef __attribute__((ext_vector_type(8)))  float  v8f;

#define WMMA_BF16(A, B, C) \
    __builtin_amdgcn_wmma_f32_16x16x32_bf16(false, (A), false, (B), (short)0, (C), false, false)

// ------------------------------------------------------------------
// Kernel 1: fused QKV projection.  y[m,n] = sum_c x[m,c] * W[n,c]
// Each wave computes one 16x64 tile (4 accumulators), K-steps of 32.
// Output stored bf16 (q pre-scaled by 1/sqrt(C)).
// ------------------------------------------------------------------
__global__ __launch_bounds__(256)
void qkv_proj_kernel(const float* __restrict__ x,
                     const float* __restrict__ Wq,
                     const float* __restrict__ Wk,
                     const float* __restrict__ Wv,
                     unsigned short* __restrict__ qbu,
                     unsigned short* __restrict__ kbu,
                     unsigned short* __restrict__ vbu,
                     int M, int C)
{
    const int lane = threadIdx.x & 31;
    const int wave = threadIdx.x >> 5;
    const int hk   = lane >> 4;    // 0 or 1: which K half-group this lane holds
    const int ml   = lane & 15;

    const int ntilesN     = C / 64;               // 16
    const int tilesPerMat = (M / 16) * ntilesN;   // 8192
    int t   = blockIdx.x * 8 + wave;
    int mat = t / tilesPerMat;
    if (mat > 2) return;
    int rem = t - mat * tilesPerMat;
    int mt  = rem / ntilesN;
    int nt  = rem - mt * ntilesN;

    const float* Wm = (mat == 0) ? Wq : (mat == 1) ? Wk : Wv;
    __bf16* outb = reinterpret_cast<__bf16*>((mat == 0) ? qbu : (mat == 1) ? kbu : vbu);
    const float scale = (mat == 0) ? 0.03125f : 1.0f;   // 1/sqrt(1024) folded into q

    const int m0 = mt * 16;
    const int n0 = nt * 64;

    v8f acc0 = {}, acc1 = {}, acc2 = {}, acc3 = {};
    const float* xr  = x  + (size_t)(m0 + ml) * C;
    const float* wr0 = Wm + (size_t)(n0 +  0 + ml) * C;
    const float* wr1 = Wm + (size_t)(n0 + 16 + ml) * C;
    const float* wr2 = Wm + (size_t)(n0 + 32 + ml) * C;
    const float* wr3 = Wm + (size_t)(n0 + 48 + ml) * C;

    for (int k0 = 0; k0 < C; k0 += 32) {
        // A fragment: lane holds M=ml, K = hk*16 .. +15
        v16bf a;
        #pragma unroll
        for (int i = 0; i < 8; ++i) {
            a[i]     = (__bf16)xr[k0 + hk * 16 + i];
            a[i + 8] = (__bf16)xr[k0 + hk * 16 + 8 + i];
        }
        // B fragments: lane holds N=ml, K = hk*16 .. +15 (contiguous in W row)
        v16bf b0, b1, b2, b3;
        #pragma unroll
        for (int i = 0; i < 16; ++i) {
            b0[i] = (__bf16)wr0[k0 + hk * 16 + i];
            b1[i] = (__bf16)wr1[k0 + hk * 16 + i];
            b2[i] = (__bf16)wr2[k0 + hk * 16 + i];
            b3[i] = (__bf16)wr3[k0 + hk * 16 + i];
        }
        acc0 = WMMA_BF16(a, b0, acc0);
        acc1 = WMMA_BF16(a, b1, acc1);
        acc2 = WMMA_BF16(a, b2, acc2);
        acc3 = WMMA_BF16(a, b3, acc3);
    }

    // C/D layout: row = r + 8*hk, col = ml (+ tile offset)
    #pragma unroll
    for (int r = 0; r < 8; ++r) {
        int row = m0 + r + 8 * hk;
        __bf16* orow = outb + (size_t)row * C + n0;
        orow[ 0 + ml] = (__bf16)(acc0[r] * scale);
        orow[16 + ml] = (__bf16)(acc1[r] * scale);
        orow[32 + ml] = (__bf16)(acc2[r] * scale);
        orow[48 + ml] = (__bf16)(acc3[r] * scale);
    }
}

// ------------------------------------------------------------------
// Kernel 2: flash attention. One workgroup (8 waves) per (b, 16-row q tile).
// Key blocks of 32. Each wave owns a 128-wide slice of d=1024:
//   - partial QK^T over its slice -> LDS reduce -> online softmax (16 thr)
//   - P (bf16, LDS) @ V-slice (LDS, transposed) into persistent f32 accums
// ------------------------------------------------------------------
__global__ __launch_bounds__(256)
void attn_kernel(const unsigned short* __restrict__ qbu,
                 const unsigned short* __restrict__ kbu,
                 const unsigned short* __restrict__ vbu,
                 float* __restrict__ out,
                 int T, int C)
{
    __shared__ __bf16 vT[1024 * 32];      // [col][key]  64 KB
    __shared__ float  Spart[8][16][32];   // per-wave partial scores, 16 KB
    __shared__ float  Ssh[16][32];        // reduced scores
    __shared__ __bf16 Psh[16][32];        // probabilities (bf16)
    __shared__ float  alpha_sh[16], mstat[16], lstat[16];

    const __bf16* qb = reinterpret_cast<const __bf16*>(qbu);
    const __bf16* kb = reinterpret_cast<const __bf16*>(kbu);
    const __bf16* vb = reinterpret_cast<const __bf16*>(vbu);

    const int tid  = threadIdx.x;
    const int lane = tid & 31;
    const int wave = tid >> 5;
    const int hk   = lane >> 4;
    const int ml   = lane & 15;

    const int b   = blockIdx.x / (T / 16);
    const int qt  = blockIdx.x % (T / 16);
    const int m0  = qt * 16;
    const int d0  = wave * 128;            // this wave's d-slice
    const size_t base = (size_t)b * T;

    // Q fragments for this wave's 128-dim slice (4 K-steps of 32)
    v16bf qf[4];
    {
        const __bf16* qr = qb + (base + m0 + ml) * C + d0;
        #pragma unroll
        for (int ks = 0; ks < 4; ++ks)
            #pragma unroll
            for (int i = 0; i < 8; ++i) {
                qf[ks][i]     = qr[ks * 32 + hk * 16 + i];
                qf[ks][i + 8] = qr[ks * 32 + hk * 16 + 8 + i];
            }
    }

    if (tid < 16) { mstat[tid] = -INFINITY; lstat[tid] = 0.f; }

    v8f acc[8];
    #pragma unroll
    for (int i = 0; i < 8; ++i) acc[i] = (v8f){};

    const int nblk = (m0 + 15) / 32 + 1;   // causal: keys 0 .. m0+15
    for (int blk = 0; blk < nblk; ++blk) {
        const int j0 = blk * 32;
        __syncthreads();   // protect vT/Psh from previous iteration readers

        // stage V^T for this key block: vT[col][key], cooperative copy
        for (int idx = tid; idx < 32 * 128; idx += 256) {
            int key = idx >> 7;
            int cu  = (idx & 127) * 8;
            const __bf16* vr = vb + (base + j0 + key) * C + cu;
            #pragma unroll
            for (int i = 0; i < 8; ++i)
                vT[(cu + i) * 32 + key] = vr[i];
        }

        // partial scores over this wave's 128-dim slice (two 16-key n-tiles)
        v8f s0 = {}, s1 = {};
        #pragma unroll
        for (int ks = 0; ks < 4; ++ks) {
            v16bf kf0, kf1;
            const __bf16* kr0 = kb + (base + j0 +  0 + ml) * C + d0 + ks * 32 + hk * 16;
            const __bf16* kr1 = kb + (base + j0 + 16 + ml) * C + d0 + ks * 32 + hk * 16;
            #pragma unroll
            for (int i = 0; i < 16; ++i) { kf0[i] = kr0[i]; kf1[i] = kr1[i]; }
            s0 = WMMA_BF16(qf[ks], kf0, s0);
            s1 = WMMA_BF16(qf[ks], kf1, s1);
        }
        #pragma unroll
        for (int r = 0; r < 8; ++r) {
            Spart[wave][r + 8 * hk][ 0 + ml] = s0[r];
            Spart[wave][r + 8 * hk][16 + ml] = s1[r];
        }
        __syncthreads();

        // cross-wave reduce + causal mask
        for (int e = tid; e < 512; e += 256) {
            int mr = e >> 5, n = e & 31;
            float s = 0.f;
            #pragma unroll
            for (int w = 0; w < 8; ++w) s += Spart[w][mr][n];
            Ssh[mr][n] = (j0 + n <= m0 + mr) ? s : -INFINITY;
        }
        __syncthreads();

        // online softmax row update (one thread per q row)
        if (tid < 16) {
            float mold = mstat[tid];
            float rmax = mold;
            #pragma unroll
            for (int n = 0; n < 32; ++n) rmax = fmaxf(rmax, Ssh[tid][n]);
            float al = expf(mold - rmax);
            float ls = lstat[tid] * al;
            #pragma unroll
            for (int n = 0; n < 32; ++n) {
                float p = expf(Ssh[tid][n] - rmax);
                Psh[tid][n] = (__bf16)p;
                ls += p;
            }
            mstat[tid] = rmax; lstat[tid] = ls; alpha_sh[tid] = al;
        }
        __syncthreads();

        // P @ V for this wave's column slice; alpha-rescale via C operand
        v16bf pf;
        #pragma unroll
        for (int i = 0; i < 8; ++i) {
            pf[i]     = Psh[ml][hk * 16 + i];
            pf[i + 8] = Psh[ml][hk * 16 + 8 + i];
        }
        float alphas[8];
        #pragma unroll
        for (int r = 0; r < 8; ++r) alphas[r] = alpha_sh[r + 8 * hk];

        #pragma unroll
        for (int nt = 0; nt < 8; ++nt) {
            v16bf vf;
            const __bf16* vc = &vT[(d0 + nt * 16 + ml) * 32 + hk * 16];
            #pragma unroll
            for (int i = 0; i < 16; ++i) vf[i] = vc[i];
            v8f cin;
            #pragma unroll
            for (int r = 0; r < 8; ++r) cin[r] = acc[nt][r] * alphas[r];
            acc[nt] = WMMA_BF16(pf, vf, cin);
        }
    }

    __syncthreads();
    float linv[8];
    #pragma unroll
    for (int r = 0; r < 8; ++r) linv[r] = 1.f / lstat[r + 8 * hk];
    #pragma unroll
    for (int nt = 0; nt < 8; ++nt)
        #pragma unroll
        for (int r = 0; r < 8; ++r)
            out[(base + m0 + r + 8 * hk) * C + d0 + nt * 16 + ml] = acc[nt][r] * linv[r];
}

// ------------------------------------------------------------------
extern "C" void kernel_launch(void* const* d_in, const int* in_sizes, int n_in,
                              void* d_out, int out_size, void* d_ws, size_t ws_size,
                              hipStream_t stream)
{
    const float* x  = (const float*)d_in[0];
    const float* Wq = (const float*)d_in[1];
    const float* Wk = (const float*)d_in[2];
    const float* Wv = (const float*)d_in[3];
    float* out = (float*)d_out;

    const int B = 4, T = 2048, C = 1024;
    const int M = B * T;                      // 8192 tokens

    // bf16 q/k/v staging in workspace (3 * 8192 * 1024 * 2 = 48 MB)
    unsigned short* qb = (unsigned short*)d_ws;
    unsigned short* kb = qb + (size_t)M * C;
    unsigned short* vb = kb + (size_t)M * C;

    const int tilesPerMat = (M / 16) * (C / 64);   // 8192
    const int totalWaves  = tilesPerMat * 3;       // 24576
    const int blocks1     = totalWaves / 8;        // 3072 blocks of 8 waves
    qkv_proj_kernel<<<blocks1, 256, 0, stream>>>(x, Wq, Wk, Wv, qb, kb, vb, M, C);

    const int blocks2 = B * (T / 16);              // 512
    attn_kernel<<<blocks2, 256, 0, stream>>>(qb, kb, vb, out, T, C);
}
